// MeanShift_65309272703420
// MI455X (gfx1250) — compile-verified, matrix operand order checked by reading, not run on hardware
//
#include <hip/hip_runtime.h>
#include <hip/hip_bf16.h>

typedef __attribute__((ext_vector_type(16))) _Float16 v16h;
typedef __attribute__((ext_vector_type(8)))  float    v8f;

#define KC   256      // K-chunk staged in LDS (f16)
#define BQ   256      // batch
#define DQ   512      // feature dim
#define HQ   4096     // hidden dim
#define KQ   128000   // queue size
#define TOPK 5

// ---------------------------------------------------------------------------
// Async global->LDS copy (CDNA5 GLOBAL_LOAD_ASYNC_TO_LDS_B128, ASYNCcnt).
// lds_off: byte address of destination within the workgroup LDS allocation.
// ---------------------------------------------------------------------------
__device__ __forceinline__ void async_copy_b128(uint32_t lds_off, uint64_t gaddr) {
  asm volatile("global_load_async_to_lds_b128 %0, %1, off"
               :: "v"(lds_off), "v"(gaddr) : "memory");
}
__device__ __forceinline__ void wait_asynccnt0() {
  asm volatile("s_wait_asynccnt 0" ::: "memory");
}

// ---------------------------------------------------------------------------
// WMMA fragment load from LDS tile laid out [row][KC] f16, row-major.
// wave32 layout (CDNA5 ISA): lane L covers row (lane&15),
// half = lane>>4; element e maps to k = (e&7) + half*8 + (e>>3)*16.
// => two contiguous 8x f16 (16B) chunks at k = ks+half*8 and ks+16+half*8.
// ---------------------------------------------------------------------------
__device__ __forceinline__ v16h load_frag(const _Float16* lds, int rowBase, int ks, int lane) {
  const int r    = rowBase + (lane & 15);
  const int half = lane >> 4;
  const _Float16* p = lds + r * KC + ks + half * 8;
  v16h f;
  uint4* fp = reinterpret_cast<uint4*>(&f);
  fp[0] = *reinterpret_cast<const uint4*>(p);
  fp[1] = *reinterpret_cast<const uint4*>(p + 16);
  return f;
}

// 8 waves per block; wave w computes a 32x64 sub-block (2x4 tiles of 16x16)
// of the 128x128 output tile.  Accumulate one KC-deep chunk.
__device__ __forceinline__ void mma_chunk(const _Float16* As, const _Float16* Bs,
                                          v8f acc[2][4], int mwOff, int nwOff, int lane) {
#pragma unroll
  for (int ks = 0; ks < KC; ks += 32) {
    v16h a[2], b[4];
#pragma unroll
    for (int i = 0; i < 2; i++) a[i] = load_frag(As, mwOff + i * 16, ks, lane);
#pragma unroll
    for (int j = 0; j < 4; j++) b[j] = load_frag(Bs, nwOff + j * 16, ks, lane);
#pragma unroll
    for (int i = 0; i < 2; i++)
#pragma unroll
      for (int j = 0; j < 4; j++)
        acc[i][j] = __builtin_amdgcn_wmma_f32_16x16x32_f16(
            false, a[i], false, b[j], (short)0, acc[i][j], false, false);
  }
}

// ---------------------------------------------------------------------------
// l2-normalize 256 rows of length 512.  One block per row, 256 threads.
// ---------------------------------------------------------------------------
__global__ void l2norm_kernel(const float* __restrict__ in, float* __restrict__ out) {
  __shared__ float red[256];
  const int b = blockIdx.x, t = threadIdx.x;
  const float x0 = in[(size_t)b * DQ + t];
  const float x1 = in[(size_t)b * DQ + 256 + t];
  red[t] = x0 * x0 + x1 * x1;
  __syncthreads();
  for (int s = 128; s > 0; s >>= 1) {
    if (t < s) red[t] += red[t + s];
    __syncthreads();
  }
  const float inv = rsqrtf(red[0]);
  out[(size_t)b * DQ + t]       = x0 * inv;
  out[(size_t)b * DQ + 256 + t] = x1 * inv;
}

// ---------------------------------------------------------------------------
// Build the effective memory bank in f16: rows < 256 come from tgt (the
// dequeue-and-enqueue), the rest from the (already l2-normalized) queue.
// Queue is read-once -> non-temporal loads; the f16 bank (131 MB) is written
// with regular hints so it stays resident in the 192 MB L2 for the GEMM.
// One float4 per thread.
// ---------------------------------------------------------------------------
__global__ void build_bank_kernel(const float* __restrict__ queue, const float* __restrict__ tgt,
                                  _Float16* __restrict__ bank) {
  const size_t gid  = (size_t)blockIdx.x * 256 + threadIdx.x;
  const size_t flat = gid * 4;                   // element index (4 | DQ, never crosses rows)
  const size_t k    = flat / DQ;
  const int    d    = (int)(flat % DQ);
  const float* src  = (k < BQ) ? (tgt + k * DQ + d) : (queue + k * DQ + d);
  const float a0 = __builtin_nontemporal_load(src + 0);
  const float a1 = __builtin_nontemporal_load(src + 1);
  const float a2 = __builtin_nontemporal_load(src + 2);
  const float a3 = __builtin_nontemporal_load(src + 3);
  _Float16 t4[4] = {(_Float16)a0, (_Float16)a1, (_Float16)a2, (_Float16)a3};
  *reinterpret_cast<uint2*>(&bank[flat]) = *reinterpret_cast<uint2*>(t4);
}

// ---------------------------------------------------------------------------
// GEMM1: H[256,4096] = X[256,512] @ W1[512,4096] + b1   (f16 WMMA, f32 accum)
// ---------------------------------------------------------------------------
__global__ void gemm1_kernel(const float* __restrict__ X, const float* __restrict__ W1,
                             const float* __restrict__ b1, float* __restrict__ Hout) {
  __shared__ _Float16 As[128 * KC];
  __shared__ _Float16 Bs[128 * KC];
  const int n0 = blockIdx.x * 128;
  const int m0 = blockIdx.y * 128;
  const int tid = threadIdx.x, lane = tid & 31, w = tid >> 5;
  const int mwOff = (w & 3) * 32, nwOff = (w >> 2) * 64;
  v8f acc[2][4] = {};
  for (int dbase = 0; dbase < DQ; dbase += KC) {
    __syncthreads();
    for (int i = tid; i < (128 * KC) / 4; i += 256) {
      const int flat = i * 4;
      const int r = flat / KC, d = flat % KC;
      const float4 v = *reinterpret_cast<const float4*>(&X[(size_t)(m0 + r) * DQ + dbase + d]);
      _Float16 t4[4] = {(_Float16)v.x, (_Float16)v.y, (_Float16)v.z, (_Float16)v.w};
      *reinterpret_cast<uint2*>(&As[r * KC + d]) = *reinterpret_cast<uint2*>(t4);
    }
    for (int i = tid; i < (128 * KC) / 4; i += 256) {
      const int flat = i * 4;
      const int d = flat >> 7, n = flat & 127;
      const float4 v = *reinterpret_cast<const float4*>(&W1[(size_t)(dbase + d) * HQ + n0 + n]);
      Bs[(n + 0) * KC + d] = (_Float16)v.x;
      Bs[(n + 1) * KC + d] = (_Float16)v.y;
      Bs[(n + 2) * KC + d] = (_Float16)v.z;
      Bs[(n + 3) * KC + d] = (_Float16)v.w;
    }
    __syncthreads();
    mma_chunk(As, Bs, acc, mwOff, nwOff, lane);
  }
#pragma unroll
  for (int i = 0; i < 2; i++)
#pragma unroll
    for (int j = 0; j < 4; j++) {
      const int m = m0 + mwOff + i * 16 + (lane >> 4) * 8;
      const int n = n0 + nwOff + j * 16 + (lane & 15);
      const float bias = b1[n];
      v8f c = acc[i][j];
#pragma unroll
      for (int e = 0; e < 8; e++) Hout[(size_t)(m + e) * HQ + n] = c[e] + bias;
    }
}

// ---------------------------------------------------------------------------
// BatchNorm batch statistics over 256 rows for each of 4096 columns.
// ---------------------------------------------------------------------------
__global__ void bn_stats_kernel(const float* __restrict__ H, float* __restrict__ mu,
                                float* __restrict__ rsig) {
  const int j = blockIdx.x * 256 + threadIdx.x;
  float s = 0.f, ss = 0.f;
  for (int b = 0; b < BQ; b++) {
    const float v = H[(size_t)b * HQ + j];
    s += v;
    ss += v * v;
  }
  const float m   = s * (1.f / BQ);
  const float var = ss * (1.f / BQ) - m * m;
  mu[j]   = m;
  rsig[j] = rsqrtf(var + 1e-5f);
}

// ---------------------------------------------------------------------------
// GEMM2: Qpre[256,512] = relu(bn(H)) @ W2[4096,512] + b2   (BN/ReLU fused in
// A-staging).
// ---------------------------------------------------------------------------
__global__ void gemm2_kernel(const float* __restrict__ H, const float* __restrict__ mu,
                             const float* __restrict__ rsig, const float* __restrict__ gamma,
                             const float* __restrict__ beta, const float* __restrict__ W2,
                             const float* __restrict__ b2, float* __restrict__ Qpre) {
  __shared__ _Float16 As[128 * KC];
  __shared__ _Float16 Bs[128 * KC];
  const int n0 = blockIdx.x * 128;
  const int m0 = blockIdx.y * 128;
  const int tid = threadIdx.x, lane = tid & 31, w = tid >> 5;
  const int mwOff = (w & 3) * 32, nwOff = (w >> 2) * 64;
  v8f acc[2][4] = {};
  for (int dbase = 0; dbase < HQ; dbase += KC) {
    __syncthreads();
    for (int i = tid; i < (128 * KC) / 4; i += 256) {
      const int flat = i * 4;
      const int r = flat / KC, d = flat % KC;
      const float4 hv = *reinterpret_cast<const float4*>(&H[(size_t)(m0 + r) * HQ + dbase + d]);
      const float4 mv = *reinterpret_cast<const float4*>(&mu[dbase + d]);
      const float4 rv = *reinterpret_cast<const float4*>(&rsig[dbase + d]);
      const float4 gv = *reinterpret_cast<const float4*>(&gamma[dbase + d]);
      const float4 bv = *reinterpret_cast<const float4*>(&beta[dbase + d]);
      float a0 = fmaxf((hv.x - mv.x) * rv.x * gv.x + bv.x, 0.f);
      float a1 = fmaxf((hv.y - mv.y) * rv.y * gv.y + bv.y, 0.f);
      float a2 = fmaxf((hv.z - mv.z) * rv.z * gv.z + bv.z, 0.f);
      float a3 = fmaxf((hv.w - mv.w) * rv.w * gv.w + bv.w, 0.f);
      _Float16 t4[4] = {(_Float16)a0, (_Float16)a1, (_Float16)a2, (_Float16)a3};
      *reinterpret_cast<uint2*>(&As[r * KC + d]) = *reinterpret_cast<uint2*>(t4);
    }
    for (int i = tid; i < (128 * KC) / 4; i += 256) {
      const int flat = i * 4;
      const int d = flat >> 7, n = flat & 127;
      const float4 v = *reinterpret_cast<const float4*>(&W2[(size_t)(dbase + d) * DQ + n0 + n]);
      Bs[(n + 0) * KC + d] = (_Float16)v.x;
      Bs[(n + 1) * KC + d] = (_Float16)v.y;
      Bs[(n + 2) * KC + d] = (_Float16)v.z;
      Bs[(n + 3) * KC + d] = (_Float16)v.w;
    }
    __syncthreads();
    mma_chunk(As, Bs, acc, mwOff, nwOff, lane);
  }
#pragma unroll
  for (int i = 0; i < 2; i++)
#pragma unroll
    for (int j = 0; j < 4; j++) {
      const int m = m0 + mwOff + i * 16 + (lane >> 4) * 8;
      const int n = n0 + nwOff + j * 16 + (lane & 15);
      const float bias = b2[n];
      v8f c = acc[i][j];
#pragma unroll
      for (int e = 0; e < 8; e++) Qpre[(size_t)(m + e) * DQ + n] = c[e] + bias;
    }
}

// ---------------------------------------------------------------------------
// Big similarity GEMM: S[256,128000] = bank[0:256] @ bank^T (bank rows 0..255
// are tgt).  A/B tiles are staged with async global->LDS B128 copies from the
// L2-resident f16 bank; scores are streamed out as f16 with NT stores so they
// never evict the bank from L2.
// ---------------------------------------------------------------------------
__global__ void dist_gemm_kernel(const _Float16* __restrict__ bank, _Float16* __restrict__ S) {
  __shared__ _Float16 As[128 * KC];
  __shared__ _Float16 Bs[128 * KC];
  const int k0 = blockIdx.x * 128;
  const int m0 = blockIdx.y * 128;
  const int tid = threadIdx.x, lane = tid & 31, w = tid >> 5;
  const int mwOff = (w & 3) * 32, nwOff = (w >> 2) * 64;
  const uint32_t asBase = (uint32_t)(uintptr_t)As;
  const uint32_t bsBase = (uint32_t)(uintptr_t)Bs;
  v8f acc[2][4] = {};
  for (int dbase = 0; dbase < DQ; dbase += KC) {
    __syncthreads();
    // 16 iterations x 2 async B128 copies per thread; LDS layout == flat f16
    // index, so lds byte offset = flat*2.
    for (int i = tid; i < (128 * KC) / 8; i += 256) {
      const int flat = i * 8;
      const int r = flat / KC, d = flat % KC;
      async_copy_b128(asBase + (uint32_t)(flat * 2),
                      (uint64_t)(bank + (size_t)(m0 + r) * DQ + dbase + d));
      async_copy_b128(bsBase + (uint32_t)(flat * 2),
                      (uint64_t)(bank + (size_t)(k0 + r) * DQ + dbase + d));
    }
    wait_asynccnt0();
    __syncthreads();
    mma_chunk(As, Bs, acc, mwOff, nwOff, lane);
  }
#pragma unroll
  for (int i = 0; i < 2; i++)
#pragma unroll
    for (int j = 0; j < 4; j++) {
      const int m = m0 + mwOff + i * 16 + (lane >> 4) * 8;
      const int n = k0 + nwOff + j * 16 + (lane & 15);
      v8f c = acc[i][j];
#pragma unroll
      for (int e = 0; e < 8; e++) {
        const _Float16 hv = (_Float16)c[e];
        __builtin_nontemporal_store(hv, &S[(size_t)(m + e) * KQ + n]);
      }
    }
}

// ---------------------------------------------------------------------------
// Per-row top-5 of similarity (== smallest distance), then recompute the 5
// needed query-branch distances + label matches.  One block per batch row.
// ---------------------------------------------------------------------------
#define TOPK_INSERT(vv, kk)                                                         \
  if ((vv) > bv4) {                                                                 \
    if ((vv) > bv0)      { bv4=bv3;bi4=bi3; bv3=bv2;bi3=bi2; bv2=bv1;bi2=bi1;       \
                           bv1=bv0;bi1=bi0; bv0=(vv);bi0=(kk); }                    \
    else if ((vv) > bv1) { bv4=bv3;bi4=bi3; bv3=bv2;bi3=bi2; bv2=bv1;bi2=bi1;       \
                           bv1=(vv);bi1=(kk); }                                     \
    else if ((vv) > bv2) { bv4=bv3;bi4=bi3; bv3=bv2;bi3=bi2; bv2=(vv);bi2=(kk); }   \
    else if ((vv) > bv3) { bv4=bv3;bi4=bi3; bv3=(vv);bi3=(kk); }                    \
    else                 { bv4=(vv);bi4=(kk); }                                     \
  }

__global__ void topk_kernel(const _Float16* __restrict__ S, const float* __restrict__ queue,
                            const float* __restrict__ tgt, const float* __restrict__ query,
                            const int* __restrict__ labels, const int* __restrict__ qlabels,
                            float* __restrict__ dqsum, float* __restrict__ match) {
  const int b = blockIdx.x, t = threadIdx.x;
  float bv0 = -1e30f, bv1 = -1e30f, bv2 = -1e30f, bv3 = -1e30f, bv4 = -1e30f;
  int   bi0 = 0, bi1 = 0, bi2 = 0, bi3 = 0, bi4 = 0;
  const _Float16* row = S + (size_t)b * KQ;
  for (int k = t; k < KQ; k += 256) {
    const _Float16 hv = __builtin_nontemporal_load(&row[k]);
    const float v = (float)hv;
    TOPK_INSERT(v, k);
  }
  __shared__ float cv[256 * TOPK];
  __shared__ int   ci[256 * TOPK];
  cv[t*TOPK+0]=bv0; cv[t*TOPK+1]=bv1; cv[t*TOPK+2]=bv2; cv[t*TOPK+3]=bv3; cv[t*TOPK+4]=bv4;
  ci[t*TOPK+0]=bi0; ci[t*TOPK+1]=bi1; ci[t*TOPK+2]=bi2; ci[t*TOPK+3]=bi3; ci[t*TOPK+4]=bi4;
  __syncthreads();
  __shared__ int sel[TOPK];
  if (t == 0) {
    float gv0=-1e30f, gv1=-1e30f, gv2=-1e30f, gv3=-1e30f, gv4=-1e30f;
    int   gi0=0, gi1=0, gi2=0, gi3=0, gi4=0;
    for (int c = 0; c < 256 * TOPK; c++) {
      const float v = cv[c];
      const int   k = ci[c];
      if (v > gv4) {
        if (v > gv0)      { gv4=gv3;gi4=gi3; gv3=gv2;gi3=gi2; gv2=gv1;gi2=gi1;
                            gv1=gv0;gi1=gi0; gv0=v;gi0=k; }
        else if (v > gv1) { gv4=gv3;gi4=gi3; gv3=gv2;gi3=gi2; gv2=gv1;gi2=gi1; gv1=v;gi1=k; }
        else if (v > gv2) { gv4=gv3;gi4=gi3; gv3=gv2;gi3=gi2; gv2=v;gi2=k; }
        else if (v > gv3) { gv4=gv3;gi4=gi3; gv3=v;gi3=k; }
        else              { gv4=v;gi4=k; }
      }
    }
    sel[0]=gi0; sel[1]=gi1; sel[2]=gi2; sel[3]=gi3; sel[4]=gi4;
  }
  __syncthreads();
  // dist_q at the 5 selected neighbors: 2 - 2 * query_b . q_eff[k]
  __shared__ float red[256];
  float total = 0.f;
  int mcount = 0;
  for (int j = 0; j < TOPK; j++) {
    const int k = sel[j];
    const float* qrow = (k < BQ) ? (tgt + (size_t)k * DQ) : (queue + (size_t)k * DQ);
    const float p = query[(size_t)b * DQ + t] * qrow[t] +
                    query[(size_t)b * DQ + 256 + t] * qrow[256 + t];
    red[t] = p;
    __syncthreads();
    for (int s = 128; s > 0; s >>= 1) {
      if (t < s) red[t] += red[t + s];
      __syncthreads();
    }
    if (t == 0) {
      total += 2.f - 2.f * red[0];
      const int lab = (k < BQ) ? labels[k] : qlabels[k];
      if (lab == labels[b]) mcount++;
    }
    __syncthreads();
  }
  if (t == 0) {
    dqsum[b] = total;
    match[b] = (float)mcount;
  }
}

// ---------------------------------------------------------------------------
// Final reduction to (loss, purity).
// ---------------------------------------------------------------------------
__global__ void finalize_kernel(const float* __restrict__ dqsum, const float* __restrict__ match,
                                float* __restrict__ out) {
  __shared__ float r1[256], r2[256];
  const int t = threadIdx.x;
  r1[t] = dqsum[t];
  r2[t] = match[t];
  __syncthreads();
  for (int s = 128; s > 0; s >>= 1) {
    if (t < s) { r1[t] += r1[t + s]; r2[t] += r2[t + s]; }
    __syncthreads();
  }
  if (t == 0) {
    out[0] = r1[0] / (float)(BQ * TOPK);   // loss
    out[1] = r2[0] / (float)(BQ * TOPK);   // purity
  }
}

// ---------------------------------------------------------------------------
extern "C" void kernel_launch(void* const* d_in, const int* in_sizes, int n_in,
                              void* d_out, int out_size, void* d_ws, size_t ws_size,
                              hipStream_t stream) {
  const float* feat_q  = (const float*)d_in[0];
  const float* cur_tgt = (const float*)d_in[1];
  const int*   labels  = (const int*)d_in[2];
  const float* queue   = (const float*)d_in[3];
  const int*   qlabels = (const int*)d_in[4];
  const float* W1      = (const float*)d_in[5];
  const float* b1      = (const float*)d_in[6];
  const float* gamma   = (const float*)d_in[7];
  const float* beta    = (const float*)d_in[8];
  const float* W2      = (const float*)d_in[9];
  const float* b2      = (const float*)d_in[10];
  float* out = (float*)d_out;

  float* ws    = (float*)d_ws;
  float* tgt   = ws;                               // 256*512      = 131072 f32
  float* qpre  = ws + 131072;                      // 256*512
  float* query = ws + 262144;                      // 256*512
  float* h     = ws + 393216;                      // 256*4096     = 1048576
  float* mu    = ws + 1441792;                     // 4096
  float* rsig  = ws + 1445888;                     // 4096
  float* dqsum = ws + 1449984;                     // 256
  float* match = ws + 1450240;                     // 256
  _Float16* s16  = (_Float16*)(ws + 1450496);      // 256*128000 f16 = 16384000 f32 slots
  _Float16* bank = (_Float16*)(ws + 17834496);     // 128000*512 f16 = 32768000 f32 slots

  // target branch + effective memory bank (f16, L2-resident)
  l2norm_kernel<<<dim3(BQ), dim3(256), 0, stream>>>(cur_tgt, tgt);
  build_bank_kernel<<<dim3((KQ * DQ / 4) / 256), dim3(256), 0, stream>>>(queue, tgt, bank);
  // query branch MLP
  gemm1_kernel<<<dim3(HQ / 128, BQ / 128), dim3(256), 0, stream>>>(feat_q, W1, b1, h);
  bn_stats_kernel<<<dim3(HQ / 256), dim3(256), 0, stream>>>(h, mu, rsig);
  gemm2_kernel<<<dim3(DQ / 128, BQ / 128), dim3(256), 0, stream>>>(h, mu, rsig, gamma, beta,
                                                                   W2, b2, qpre);
  l2norm_kernel<<<dim3(BQ), dim3(256), 0, stream>>>(qpre, query);
  // target-branch similarity against the bank (async-LDS staged WMMA)
  dist_gemm_kernel<<<dim3(KQ / 128, BQ / 128), dim3(256), 0, stream>>>(bank, s16);
  // top-5 neighbors + query-branch distances at those neighbors
  topk_kernel<<<dim3(BQ), dim3(256), 0, stream>>>(s16, queue, tgt, query, labels, qlabels,
                                                  dqsum, match);
  finalize_kernel<<<dim3(1), dim3(256), 0, stream>>>(dqsum, match, out);
}
